// K_Means_77627238908056
// MI455X (gfx1250) — matrix operand order repaired and checked by reading, not compiled, not run on hardware
//
#include <hip/hip_runtime.h>

typedef __attribute__((ext_vector_type(16))) __bf16          v16bf;
typedef __attribute__((ext_vector_type(16))) unsigned short  v16u;
typedef __attribute__((ext_vector_type(8)))  float           v8f;

#define N_PTS 131072
#define D_DIM 128
#define K_CTR 512
#define ROWS_PER_BLOCK 64
#define XL_STRIDE 132   // padded LDS row stride (floats) to avoid bank conflicts

__device__ __forceinline__ unsigned short f2bf(float f) {
  // round-to-nearest-even fp32 -> bf16 (inputs are finite normals here)
  unsigned int u = __float_as_uint(f);
  u += 0x7FFFu + ((u >> 16) & 1u);
  return (unsigned short)(u >> 16);
}
__device__ __forceinline__ float bf2f(unsigned short h) {
  return __uint_as_float(((unsigned int)h) << 16);
}

__global__ void zero_ws_kernel(float* __restrict__ p, int n) {
  int i = blockIdx.x * blockDim.x + threadIdx.x;
  if (i < n) p[i] = 0.0f;
}

// One block per centroid: emit bf16 hi/lo decomposition + ||c||^2.
__global__ __launch_bounds__(D_DIM)
void prep_centroids_kernel(const float* __restrict__ cent,
                           unsigned short* __restrict__ chi,
                           unsigned short* __restrict__ clo,
                           float* __restrict__ c2) {
  int k = blockIdx.x;
  int d = threadIdx.x;
  float c = cent[(size_t)k * D_DIM + d];
  unsigned short h = f2bf(c);
  unsigned short l = f2bf(c - bf2f(h));
  chi[(size_t)k * D_DIM + d] = h;
  clo[(size_t)k * D_DIM + d] = l;

  __shared__ float red[D_DIM];
  red[d] = c * c;
  __syncthreads();
  for (int s = D_DIM / 2; s > 0; s >>= 1) {
    if (d < s) red[d] += red[d + s];
    __syncthreads();
  }
  if (d == 0) c2[k] = red[0];
}

// 128 threads = 4 waves; each wave owns 16 rows of x; block owns 64 rows.
// Each wave computes dot(x_rows, all 512 centroids) via bf16 split WMMA,
// tracks argmin of (||c||^2 - 2 x.c), then atomically accumulates sums/counts.
__global__ __launch_bounds__(128)
void kmeans_assign_accum_kernel(const float* __restrict__ x,
                                const unsigned short* __restrict__ chi,
                                const unsigned short* __restrict__ clo,
                                const float* __restrict__ c2g,
                                float* __restrict__ sums,
                                float* __restrict__ counts) {
  __shared__ float xl[ROWS_PER_BLOCK * XL_STRIDE];
  __shared__ float c2l[K_CTR];
  __shared__ int   asg[ROWS_PER_BLOCK];

  const int tid  = threadIdx.x;
  const int wv   = tid >> 5;      // wave id 0..3
  const int lane = tid & 31;
  const int g    = lane >> 4;     // half-wave group (0/1)
  const int nl   = lane & 15;     // lane-in-half = column index
  const int m0   = blockIdx.x * ROWS_PER_BLOCK;

  // ---- stage this wave's 16 rows of x into LDS (float4 coalesced) ----
  {
    const float4* src = (const float4*)(x + (size_t)(m0 + wv * 16) * D_DIM);
    for (int i = lane; i < 16 * (D_DIM / 4); i += 32) {
      int r  = i >> 5;            // 32 float4 per row
      int c4 = i & 31;
      float4 v = src[(size_t)r * (D_DIM / 4) + c4];
      *(float4*)&xl[(wv * 16 + r) * XL_STRIDE + c4 * 4] = v;
    }
  }
  for (int i = tid; i < K_CTR; i += 128) c2l[i] = c2g[i];
  __syncthreads();

  // ---- build A fragments (16x32 bf16 per 32-wide d-chunk), hi+lo split ----
  // Lane layout (16-bit A 16x32): lane m=nl; e=0..15 -> K=16g+e
  const float* xrow = &xl[(wv * 16 + nl) * XL_STRIDE];
  v16u a_hi[4], a_lo[4];
#pragma unroll
  for (int c = 0; c < 4; ++c) {
    const int d0 = 32 * c;
#pragma unroll
    for (int e = 0; e < 8; ++e) {
      float f0 = xrow[d0 + 16 * g + e];
      unsigned short h0 = f2bf(f0);
      a_hi[c][e] = h0;
      a_lo[c][e] = f2bf(f0 - bf2f(h0));
      float f1 = xrow[d0 + 16 * g + 8 + e];
      unsigned short h1 = f2bf(f1);
      a_hi[c][8 + e] = h1;
      a_lo[c][8 + e] = f2bf(f1 - bf2f(h1));
    }
  }

  // ---- sweep all 512 centroids in 16-wide tiles ----
  float best[8];
  int   bidx[8];
#pragma unroll
  for (int r = 0; r < 8; ++r) { best[r] = 3.4e38f; bidx[r] = 0; }

  for (int t = 0; t < K_CTR / 16; ++t) {
    const int n0 = t * 16;
    // B layout (16-bit B 32x16): lane col n=nl; e=0..15 -> K=16g+e  => 32B contiguous
    const size_t brow = (size_t)(n0 + nl) * D_DIM + 16 * g;
    v8f acc = {0.f, 0.f, 0.f, 0.f, 0.f, 0.f, 0.f, 0.f};
#pragma unroll
    for (int c = 0; c < 4; ++c) {
      const int d0 = 32 * c;
      v16u bhi = *(const v16u*)(chi + brow + d0);
      v16u blo = *(const v16u*)(clo + brow + d0);
      acc = __builtin_amdgcn_wmma_f32_16x16x32_bf16(
          false, __builtin_bit_cast(v16bf, a_hi[c]),
          false, __builtin_bit_cast(v16bf, bhi), (short)0, acc, false, false);
      acc = __builtin_amdgcn_wmma_f32_16x16x32_bf16(
          false, __builtin_bit_cast(v16bf, a_lo[c]),
          false, __builtin_bit_cast(v16bf, bhi), (short)0, acc, false, false);
      acc = __builtin_amdgcn_wmma_f32_16x16x32_bf16(
          false, __builtin_bit_cast(v16bf, a_hi[c]),
          false, __builtin_bit_cast(v16bf, blo), (short)0, acc, false, false);
    }
    const float c2n  = c2l[n0 + nl];
    const int   nidx = n0 + nl;
    // C/D layout: element r of v8f, lanes 0-15 -> row r, lanes 16-31 -> row r+8
#pragma unroll
    for (int r = 0; r < 8; ++r) {
      float s = c2n - 2.0f * acc[r];
      if (s < best[r]) { best[r] = s; bidx[r] = nidx; }
    }
  }

  // ---- argmin reduce across the 16 columns of each half-wave ----
#pragma unroll
  for (int r = 0; r < 8; ++r) {
    float b  = best[r];
    int   bi = bidx[r];
#pragma unroll
    for (int off = 1; off < 16; off <<= 1) {
      float ob  = __shfl_xor(b, off, 32);
      int   obi = __shfl_xor(bi, off, 32);
      if (ob < b || (ob == b && obi < bi)) { b = ob; bi = obi; }
    }
    if (nl == 0) asg[wv * 16 + r + 8 * g] = bi;  // row = r + 8*g
  }
  __syncthreads();

  // ---- segment-sum accumulation (fp32 from LDS, global float atomics) ----
  const float* xbase = &xl[(wv * 16) * XL_STRIDE];
  for (int row = 0; row < 16; ++row) {
    int a = asg[wv * 16 + row];
    float* srow = sums + (size_t)a * D_DIM;
#pragma unroll
    for (int j = 0; j < 4; ++j) {
      int d = lane + 32 * j;
      atomicAdd(&srow[d], xbase[row * XL_STRIDE + d]);
    }
    if (lane == 0) atomicAdd(&counts[a], 1.0f);
  }
}

__global__ void finalize_kernel(const float* __restrict__ sums,
                                const float* __restrict__ counts,
                                float* __restrict__ out) {
  int i = blockIdx.x * blockDim.x + threadIdx.x;
  if (i < K_CTR * D_DIM) out[i] = sums[i] / counts[i >> 7];  // 0/0 -> NaN (empty cluster)
}

extern "C" void kernel_launch(void* const* d_in, const int* in_sizes, int n_in,
                              void* d_out, int out_size, void* d_ws, size_t ws_size,
                              hipStream_t stream) {
  (void)in_sizes; (void)n_in; (void)out_size; (void)ws_size;
  const float* x    = (const float*)d_in[0];   // [N, D] fp32
  const float* cent = (const float*)d_in[1];   // [K, D] fp32
  float* out = (float*)d_out;                  // [K, D] fp32

  // workspace layout
  float* sums   = (float*)d_ws;                         // K*D floats
  float* counts = sums + (size_t)K_CTR * D_DIM;         // K floats
  float* c2     = counts + K_CTR;                       // K floats
  unsigned short* chi = (unsigned short*)(c2 + K_CTR);  // K*D bf16 (hi)
  unsigned short* clo = chi + (size_t)K_CTR * D_DIM;    // K*D bf16 (lo)

  const int nz = K_CTR * D_DIM + 2 * K_CTR;
  zero_ws_kernel<<<(nz + 255) / 256, 256, 0, stream>>>(sums, nz);
  prep_centroids_kernel<<<K_CTR, D_DIM, 0, stream>>>(cent, chi, clo, c2);
  kmeans_assign_accum_kernel<<<N_PTS / ROWS_PER_BLOCK, 128, 0, stream>>>(
      x, chi, clo, c2, sums, counts);
  finalize_kernel<<<(K_CTR * D_DIM + 255) / 256, 256, 0, stream>>>(sums, counts, out);
}
